// AydinMoEUltra_81827716923804
// MI455X (gfx1250) — compile-verified, matrix operand order checked by reading, not run on hardware
//
#include <hip/hip_runtime.h>
#include <hip/hip_bf16.h>
#include <math.h>

#define T_TOK   8192
#define H_DIM   1024
#define I_DIM   4096
#define E_NUM   8
#define M_TILE  16
#define IC_CHUNK 256        // I-chunk per block iteration (8 waves x 2 n-tiles x 16)
#define FB_CHUNK 128        // fallback kernel chunk
#define XPAD    8
#define FRAG    512         // bf16 elements per packed 32x16 B tile (32 lanes * 16)

typedef __attribute__((ext_vector_type(16))) __bf16 v16bf;
typedef __attribute__((ext_vector_type(8)))  float  v8f;

// float -> bf16, round-to-nearest-even
__device__ __forceinline__ __bf16 f2bf(float f) {
    union { float f; unsigned u; } v; v.f = f;
    unsigned r = (v.u + 0x7fffu + ((v.u >> 16) & 1u)) >> 16;
    union { unsigned short s; __bf16 b; } o; o.s = (unsigned short)r;
    return o.b;
}

__device__ __forceinline__ float fast_silu(float h) {
    return h * __builtin_amdgcn_rcpf(1.0f + __expf(-h));
}

// A-fragment (16x32 bf16 MxK): two 16B LDS loads from a per-lane row pointer.
__device__ __forceinline__ v16bf load_a16(const __bf16* p) {
    union { int4 q[2]; v16bf v; } u;
    u.q[0] = *(const int4*)p;
    u.q[1] = *(const int4*)(p + 16);
    return u.v;
}

// B-fragment from pre-packed weights: 32 contiguous bytes per lane.
__device__ __forceinline__ v16bf load_b_packed(const __bf16* p) {
    union { int4 q[2]; v16bf v; } u;
    const int4* q = (const int4*)p;
    u.q[0] = q[0]; u.q[1] = q[1];
    return u.v;
}

// B-fragment straight from fp32 weights (fallback path, converts on the fly).
__device__ __forceinline__ v16bf load_b_f32(const float* __restrict__ W, int ld,
                                            int k0, int n0, int lane) {
    const int n    = n0 + (lane & 15);
    const int koff = k0 + ((lane & 16) ? 8 : 0);
    const float* p = W + (size_t)koff * ld + n;
    v16bf b;
#pragma unroll
    for (int i = 0; i < 8; ++i) {
        b[i]     = f2bf(p[(size_t)i * ld]);
        b[i + 8] = f2bf(p[(size_t)(i + 16) * ld]);
    }
    return b;
}

// ---------------- Phase 0: pack fp32 weights into WMMA B-fragment order --------------
// pW1 layout: [e][n_tile(256)][k_tile(32)][lane(32)][16 bf16]
__global__ void pack_w1_kernel(const float* __restrict__ W1, __bf16* __restrict__ pW1) {
    const size_t gid  = (size_t)blockIdx.x * blockDim.x + threadIdx.x;
    const int    lane = (int)(gid & 31);
    const size_t tile = gid >> 5;               // (e*256 + ntile)*32 + kt
    const int    kt    = (int)(tile & 31);
    const size_t t2    = tile >> 5;
    const int    ntile = (int)(t2 & 255);
    const int    e     = (int)(t2 >> 8);
    const int n = ntile * 16 + (lane & 15);
    const int k = kt * 32 + ((lane & 16) ? 8 : 0);
    const float* src = W1 + ((size_t)e * H_DIM + k) * I_DIM + n;
    __bf16* dst = pW1 + tile * FRAG + (size_t)lane * 16;
#pragma unroll
    for (int i = 0; i < 8; ++i) {
        dst[i]     = f2bf(src[(size_t)i * I_DIM]);
        dst[i + 8] = f2bf(src[(size_t)(i + 16) * I_DIM]);
    }
}

// pW2 layout: [e][k_tile(128)][n_tile(64)][lane(32)][16 bf16]
__global__ void pack_w2_kernel(const float* __restrict__ W2, __bf16* __restrict__ pW2) {
    const size_t gid  = (size_t)blockIdx.x * blockDim.x + threadIdx.x;
    const int    lane = (int)(gid & 31);
    const size_t tile = gid >> 5;               // (e*128 + kt)*64 + ntile
    const int    ntile = (int)(tile & 63);
    const size_t t2    = tile >> 6;
    const int    kt    = (int)(t2 & 127);
    const int    e     = (int)(t2 >> 7);
    const int n = ntile * 16 + (lane & 15);
    const int k = kt * 32 + ((lane & 16) ? 8 : 0);
    const float* src = W2 + ((size_t)e * I_DIM + k) * H_DIM + n;
    __bf16* dst = pW2 + tile * FRAG + (size_t)lane * 16;
#pragma unroll
    for (int i = 0; i < 8; ++i) {
        dst[i]     = f2bf(src[(size_t)i * H_DIM]);
        dst[i + 8] = f2bf(src[(size_t)(i + 16) * H_DIM]);
    }
}

// ---------------- Phase 1: router (logits -> top-2 -> compacted expert lists) --------
__global__ void moe_router_kernel(const float* __restrict__ x, const float* __restrict__ Wr,
                                  int* __restrict__ counts, int* __restrict__ idx,
                                  float* __restrict__ wgt) {
    const int gid  = blockIdx.x * blockDim.x + threadIdx.x;
    const int tok  = gid >> 5;
    const int lane = gid & 31;
    if (tok >= T_TOK) return;

    const float* xr = x + (size_t)tok * H_DIM;
    float acc[E_NUM] = {};
    for (int h = lane; h < H_DIM; h += 32) {
        const float xv = xr[h];
        const float* wr = Wr + (size_t)h * E_NUM;
#pragma unroll
        for (int e = 0; e < E_NUM; ++e) acc[e] += xv * wr[e];
    }
#pragma unroll
    for (int e = 0; e < E_NUM; ++e) {
#pragma unroll
        for (int m = 16; m >= 1; m >>= 1) acc[e] += __shfl_xor(acc[e], m, 32);
    }
    if (lane == 0) {
        int   e1 = 0; float l1 = acc[0];
#pragma unroll
        for (int e = 1; e < E_NUM; ++e) if (acc[e] > l1) { l1 = acc[e]; e1 = e; }
        int   e2 = (e1 == 0) ? 1 : 0; float l2 = acc[e2];
#pragma unroll
        for (int e = 0; e < E_NUM; ++e) if (e != e1 && acc[e] > l2) { l2 = acc[e]; e2 = e; }
        const float w1 = 1.0f / (1.0f + __expf(l2 - l1));   // top-2 softmax renorm
        const float w2 = 1.0f - w1;
        int p = atomicAdd(&counts[e1], 1);
        idx[(size_t)e1 * T_TOK + p] = tok;  wgt[(size_t)e1 * T_TOK + p] = w1;
        p = atomicAdd(&counts[e2], 1);
        idx[(size_t)e2 * T_TOK + p] = tok;  wgt[(size_t)e2 * T_TOK + p] = w2;
    }
}

// ---------------- Phase 2 (fast): fused expert FFN on pre-packed bf16 weights --------
__global__ __launch_bounds__(256, 1)
void moe_ffn_packed_kernel(const float* __restrict__ x,
                           const __bf16* __restrict__ pW1, const float* __restrict__ b1,
                           const __bf16* __restrict__ pW2, const float* __restrict__ b2,
                           const int* __restrict__ counts, const int* __restrict__ idx,
                           const float* __restrict__ wgt, float* __restrict__ out) {
    const int e    = blockIdx.y;
    const int cnt  = counts[e];
    const int base = blockIdx.x * M_TILE;
    if (base >= cnt) return;

    __shared__ __bf16 sX[M_TILE][H_DIM + XPAD];
    __shared__ __bf16 sH[2][M_TILE][IC_CHUNK + XPAD];   // double-buffered intermediate
    __shared__ int    sTok[M_TILE];
    __shared__ float  sWgt[M_TILE];

    const int tid = threadIdx.x;
    if (tid < M_TILE) {
        const int pos = base + tid;
        if (pos < cnt) {
            sTok[tid] = idx[(size_t)e * T_TOK + pos];
            sWgt[tid] = wgt[(size_t)e * T_TOK + pos];
        } else {
            sTok[tid] = idx[(size_t)e * T_TOK + base];
            sWgt[tid] = 0.0f;
        }
    }
    __syncthreads();

    {   // stage gathered X tile to LDS as bf16 (float4 global reads)
        const int r  = tid >> 4;
        const int c0 = (tid & 15) * 64;
        const float* src = x + (size_t)sTok[r] * H_DIM + c0;
        __bf16* dst = &sX[r][c0];
#pragma unroll 4
        for (int i = 0; i < 16; ++i) {
            float4 v = ((const float4*)src)[i];
            dst[4 * i + 0] = f2bf(v.x); dst[4 * i + 1] = f2bf(v.y);
            dst[4 * i + 2] = f2bf(v.z); dst[4 * i + 3] = f2bf(v.w);
        }
    }
    __syncthreads();

    const int wv   = tid >> 5;
    const int lane = tid & 31;
    const int mh   = (lane & 16) ? 8 : 0;
    const int nl   = lane & 15;
    const int aoff = (lane & 16) ? 8 : 0;
    const __bf16* aX = &sX[nl][aoff];

    v8f accO[8] = {};
    int buf = 0;

    for (int ic = 0; ic < I_DIM; ic += IC_CHUNK, buf ^= 1) {
        // ---- GEMM1: two adjacent 16-wide h tiles per wave; A reused for both ------
        v8f accH0 = {}, accH1 = {};
        const __bf16* b1p = pW1 + (((size_t)e * 256 + (ic >> 4) + wv * 2) * 32) * FRAG
                                + (size_t)lane * 16;
        const __bf16* b2p = pW2 + ((((size_t)e * 128 + (ic >> 5)) * 64 + wv * 8) * FRAG)
                                + (size_t)lane * 16;
        if (ic + IC_CHUNK < I_DIM) {   // pull next chunk's B streams toward the caches
            __builtin_prefetch(b1p + (size_t)(IC_CHUNK / 16) * 32 * FRAG, 0, 2);
            __builtin_prefetch(b2p + (size_t)(IC_CHUNK / 32) * 64 * FRAG, 0, 2);
        }
#pragma unroll 4
        for (int kt = 0; kt < 32; ++kt) {
            v16bf a  = load_a16(aX + kt * 32);
            v16bf bA = load_b_packed(b1p + (size_t)kt * FRAG);
            v16bf bB = load_b_packed(b1p + (size_t)(32 + kt) * FRAG);
            accH0 = __builtin_amdgcn_wmma_f32_16x16x32_bf16(
                        false, a, false, bA, (short)0, accH0, false, false);
            accH1 = __builtin_amdgcn_wmma_f32_16x16x32_bf16(
                        false, a, false, bB, (short)0, accH1, false, false);
        }
        const int   c0   = wv * 32 + nl;
        const float b1v0 = b1[(size_t)e * I_DIM + ic + c0];
        const float b1v1 = b1[(size_t)e * I_DIM + ic + c0 + 16];
#pragma unroll
        for (int j = 0; j < 8; ++j) {
            sH[buf][mh + j][c0]      = f2bf(fast_silu(accH0[j] + b1v0));
            sH[buf][mh + j][c0 + 16] = f2bf(fast_silu(accH1[j] + b1v1));
        }
        __syncthreads();   // single barrier per chunk (double buffering covers WAR)

        // ---- GEMM2: accO += h_chunk @ W2[ic..ic+256, wv*128..+128) ---------------
        const __bf16* aH = &sH[buf][nl][aoff];
#pragma unroll
        for (int kkt = 0; kkt < IC_CHUNK / 32; ++kkt) {
            v16bf a = load_a16(aH + kkt * 32);
            const __bf16* bk = b2p + (size_t)kkt * 64 * FRAG;
#pragma unroll
            for (int nt = 0; nt < 8; ++nt) {
                v16bf b = load_b_packed(bk + (size_t)nt * FRAG);
                accO[nt] = __builtin_amdgcn_wmma_f32_16x16x32_bf16(
                               false, a, false, b, (short)0, accO[nt], false, false);
            }
        }
    }

#pragma unroll
    for (int nt = 0; nt < 8; ++nt) {
        const int n = wv * 128 + nt * 16 + nl;
        const float b2v = b2[(size_t)e * H_DIM + n];
#pragma unroll
        for (int j = 0; j < 8; ++j) {
            const int m = mh + j;
            atomicAdd(out + (size_t)sTok[m] * H_DIM + n, (accO[nt][j] + b2v) * sWgt[m]);
        }
    }
}

// ---------------- Phase 2 (fallback): on-the-fly conversion (small workspace) --------
__global__ __launch_bounds__(256, 1)
void moe_ffn_kernel(const float* __restrict__ x,
                    const float* __restrict__ W1, const float* __restrict__ b1,
                    const float* __restrict__ W2, const float* __restrict__ b2,
                    const int* __restrict__ counts, const int* __restrict__ idx,
                    const float* __restrict__ wgt, float* __restrict__ out) {
    const int e    = blockIdx.y;
    const int cnt  = counts[e];
    const int base = blockIdx.x * M_TILE;
    if (base >= cnt) return;

    __shared__ __bf16 sX[M_TILE][H_DIM + XPAD];
    __shared__ __bf16 sH[M_TILE][FB_CHUNK + XPAD];
    __shared__ int    sTok[M_TILE];
    __shared__ float  sWgt[M_TILE];

    const int tid = threadIdx.x;
    if (tid < M_TILE) {
        const int pos = base + tid;
        if (pos < cnt) {
            sTok[tid] = idx[(size_t)e * T_TOK + pos];
            sWgt[tid] = wgt[(size_t)e * T_TOK + pos];
        } else {
            sTok[tid] = idx[(size_t)e * T_TOK + base];
            sWgt[tid] = 0.0f;
        }
    }
    __syncthreads();
    {
        const int r  = tid >> 4;
        const int c0 = (tid & 15) * 64;
        const float* src = x + (size_t)sTok[r] * H_DIM + c0;
        __bf16* dst = &sX[r][c0];
#pragma unroll 4
        for (int i = 0; i < 16; ++i) {
            float4 v = ((const float4*)src)[i];
            dst[4 * i + 0] = f2bf(v.x); dst[4 * i + 1] = f2bf(v.y);
            dst[4 * i + 2] = f2bf(v.z); dst[4 * i + 3] = f2bf(v.w);
        }
    }
    __syncthreads();

    const int wv   = tid >> 5;
    const int lane = tid & 31;
    const int mh   = (lane & 16) ? 8 : 0;
    const int nl   = lane & 15;
    const __bf16* aX = &sX[nl][(lane & 16) ? 8 : 0];
    const __bf16* aH = &sH[nl][(lane & 16) ? 8 : 0];
    const float* W1e = W1 + (size_t)e * H_DIM * I_DIM;
    const float* W2e = W2 + (size_t)e * I_DIM * H_DIM;

    v8f accO[8] = {};
    for (int ic = 0; ic < I_DIM; ic += FB_CHUNK) {
        v8f accH = {};
        for (int kh = 0; kh < H_DIM; kh += 32) {
            v16bf a = load_a16(aX + kh);
            v16bf b = load_b_f32(W1e, I_DIM, kh, ic + wv * 16, lane);
            accH = __builtin_amdgcn_wmma_f32_16x16x32_bf16(
                       false, a, false, b, (short)0, accH, false, false);
        }
        const float b1v = b1[(size_t)e * I_DIM + ic + wv * 16 + nl];
        __syncthreads();
#pragma unroll
        for (int j = 0; j < 8; ++j)
            sH[mh + j][wv * 16 + nl] = f2bf(fast_silu(accH[j] + b1v));
        __syncthreads();
#pragma unroll
        for (int kk = 0; kk < FB_CHUNK; kk += 32) {
            v16bf a = load_a16(aH + kk);
#pragma unroll
            for (int nt = 0; nt < 8; ++nt) {
                v16bf b = load_b_f32(W2e, H_DIM, ic + kk, wv * 128 + nt * 16, lane);
                accO[nt] = __builtin_amdgcn_wmma_f32_16x16x32_bf16(
                               false, a, false, b, (short)0, accO[nt], false, false);
            }
        }
    }
#pragma unroll
    for (int nt = 0; nt < 8; ++nt) {
        const int n = wv * 128 + nt * 16 + nl;
        const float b2v = b2[(size_t)e * H_DIM + n];
#pragma unroll
        for (int j = 0; j < 8; ++j) {
            const int m = mh + j;
            atomicAdd(out + (size_t)sTok[m] * H_DIM + n, (accO[nt][j] + b2v) * sWgt[m]);
        }
    }
}

extern "C" void kernel_launch(void* const* d_in, const int* in_sizes, int n_in,
                              void* d_out, int out_size, void* d_ws, size_t ws_size,
                              hipStream_t stream) {
    (void)in_sizes; (void)n_in;
    const float* x  = (const float*)d_in[0];
    const float* Wr = (const float*)d_in[1];
    const float* W1 = (const float*)d_in[2];
    const float* b1 = (const float*)d_in[3];
    const float* W2 = (const float*)d_in[4];
    const float* b2 = (const float*)d_in[5];
    float* out = (float*)d_out;

    char*  ws     = (char*)d_ws;
    int*   counts = (int*)ws;
    int*   idx    = (int*)(ws + 256);
    float* wgt    = (float*)(ws + 256 + (size_t)E_NUM * T_TOK * sizeof(int));

    size_t routeBytes = 256 + (size_t)E_NUM * T_TOK * (sizeof(int) + sizeof(float));
    size_t packOff    = (routeBytes + 255) & ~(size_t)255;
    size_t w1Tiles    = (size_t)E_NUM * (I_DIM / 16) * (H_DIM / 32);   // 65536
    size_t w2Tiles    = (size_t)E_NUM * (I_DIM / 32) * (H_DIM / 16);   // 65536
    size_t w1Bytes    = w1Tiles * FRAG * sizeof(__bf16);               // 64 MB
    size_t w2Bytes    = w2Tiles * FRAG * sizeof(__bf16);               // 64 MB
    bool   packed     = ws_size >= packOff + w1Bytes + w2Bytes;

    hipMemsetAsync(counts, 0, 256, stream);
    hipMemsetAsync(out, 0, (size_t)out_size * sizeof(float), stream);

    moe_router_kernel<<<(T_TOK * 32) / 256, 256, 0, stream>>>(x, Wr, counts, idx, wgt);

    dim3 grid(T_TOK / M_TILE, E_NUM);
    if (packed) {
        __bf16* pW1 = (__bf16*)(ws + packOff);
        __bf16* pW2 = (__bf16*)(ws + packOff + w1Bytes);
        pack_w1_kernel<<<(unsigned)(w1Tiles * 32 / 256), 256, 0, stream>>>(W1, pW1);
        pack_w2_kernel<<<(unsigned)(w2Tiles * 32 / 256), 256, 0, stream>>>(W2, pW2);
        moe_ffn_packed_kernel<<<grid, 256, 0, stream>>>(x, pW1, b1, pW2, b2,
                                                        counts, idx, wgt, out);
    } else {
        moe_ffn_kernel<<<grid, 256, 0, stream>>>(x, W1, b1, W2, b2,
                                                 counts, idx, wgt, out);
    }
}
